// HandWritingSynthesis_54863912239914
// MI455X (gfx1250) — compile-verified, matrix operand order checked by reading, not run on hardware
//
#include <hip/hip_runtime.h>
#include <hip/hip_bf16.h>
#include <math.h>

// Problem constants (reference: H=400, K=10, V=61, U=64, B=32, T=600, NOUT=121)
#define HH   400
#define KK   10
#define VV   61
#define UU   64
#define BB   32
#define TT   600
#define NOUTC 121
#define KP1  480          // [x(3)+w(61)=64][h 400 ->416 pad] ; also in2/in3 pitch
#define KPR  416          // h padded to mult of 32
#define NZ   1600         // 4*H gate width
#define KPM  1248         // [lstm1 416][lstm2 416][lstm3 416]
#define NYM  128          // 121 -> 128 padded
#define MROWS (TT*BB)     // 19200

typedef __attribute__((ext_vector_type(16))) __bf16 v16bf;
typedef __attribute__((ext_vector_type(8)))  float  v8f;

#define LOG2E 1.4426950408889634f

#if __has_builtin(__builtin_amdgcn_exp2f)
#define FEXP2(x) __builtin_amdgcn_exp2f(x)
#else
#define FEXP2(x) exp2f(x)
#endif
#if __has_builtin(__builtin_amdgcn_rcpf)
#define FRCP(x) __builtin_amdgcn_rcpf(x)
#else
#define FRCP(x) (1.0f/(x))
#endif

static __device__ __forceinline__ float fexpn(float x){ return FEXP2(x * LOG2E); }
static __device__ __forceinline__ float sigf(float x){
  return FRCP(1.0f + FEXP2(x * (-LOG2E)));          // branch-free sigmoid
}
static __device__ __forceinline__ float ftanh(float x){
#if __has_builtin(__builtin_amdgcn_tanhf)
  return __builtin_amdgcn_tanhf(x);                  // native V_TANH (TRANS op)
#else
  // branch-free: tanh(x) = 2*sigmoid(2x) - 1
  return 2.0f * FRCP(1.0f + FEXP2(x * (-2.0f*LOG2E))) - 1.0f;
#endif
}

static __device__ __forceinline__ unsigned short f2bf(float f){
  unsigned u = __float_as_uint(f);
  u += 0x7FFFu + ((u >> 16) & 1u);      // round-to-nearest-even
  return (unsigned short)(u >> 16);
}
static __device__ __forceinline__ float bf2f(unsigned short h){
  return __uint_as_float(((unsigned)h) << 16);
}

// A-fragment (16x32 bf16, wave32 layout): lane<16 -> M=lane, K halves 0..7 & 16..23;
// lane>=16 -> M=lane-16, K halves 8..15 & 24..31.  Source row-major [M][pitch].
static __device__ __forceinline__ v16bf load_afrag(const unsigned short* A, int pitch,
                                                   int mbase, int kbase, int lane){
  int row = mbase + (lane & 15);
  int kh  = (lane >> 4) * 8;
  const unsigned short* base = A + (size_t)row * pitch + kbase + kh;
  union { v16bf v; uint4 q[2]; } r;
  r.q[0] = *(const uint4*)(base);
  r.q[1] = *(const uint4*)(base + 16);
  return r.v;
}
// B-fragment (32x16 bf16) from pre-swizzled weights: tile index relative to base,
// 1024B/tile, lane's 16 halves contiguous -> two b128 loads.
static __device__ __forceinline__ v16bf load_bfrag(const unsigned short* Bsw, size_t tile, int lane){
  union { v16bf v; uint4 q[2]; } r;
  const uint4* p = (const uint4*)(Bsw + tile * 512 + (size_t)lane * 16);
  r.q[0] = p[0]; r.q[1] = p[1];
  return r.v;
}
static __device__ __forceinline__ v8f wmma_bf16(v16bf a, v16bf b, v8f c){
  return __builtin_amdgcn_wmma_f32_16x16x32_bf16(false, a, false, b, (short)0, c, false, false);
}

// ---------------- weight pre-swizzle (fp32 -> bf16, WMMA B-fragment order) ----------------
__global__ void hw_swizzleB(unsigned short* __restrict__ dst,
                            const float* __restrict__ src0, int K0,
                            const float* __restrict__ src1, int K1,
                            int Ncols, int KT, int NT){
  int j = blockIdx.x * 256 + threadIdx.x;
  int total = KT * NT * 32;
  if (j >= total) return;
  int lane = j & 31, tile = j >> 5;
  int kt = tile % KT, nt = tile / KT;
  int n = nt * 16 + (lane & 15);
  int kh = (lane >> 4) * 8;
  unsigned short* d = dst + (size_t)tile * 512 + (size_t)lane * 16;
  #pragma unroll
  for (int e = 0; e < 16; ++e){
    int k = kt * 32 + ((e < 8) ? (kh + e) : (16 + kh + (e - 8)));
    float v = 0.0f;
    if (n < Ncols){
      if (k < K0) v = src0[(size_t)k * Ncols + n];
      else if (k < K0 + K1) v = src1[(size_t)(k - K0) * Ncols + n];
    }
    d[e] = f2bf(v);
  }
}
// Wm (1200x121) -> 1248x128 with 416-row segments (rows 400..415 of each segment zero)
__global__ void hw_swizzleWm(unsigned short* __restrict__ dst, const float* __restrict__ Wm){
  int j = blockIdx.x * 256 + threadIdx.x;
  const int KT = 39, NT = 8;
  if (j >= KT * NT * 32) return;
  int lane = j & 31, tile = j >> 5;
  int kt = tile % KT, nt = tile / KT;
  int n = nt * 16 + (lane & 15);
  int kh = (lane >> 4) * 8;
  unsigned short* d = dst + (size_t)tile * 512 + (size_t)lane * 16;
  #pragma unroll
  for (int e = 0; e < 16; ++e){
    int k = kt * 32 + ((e < 8) ? (kh + e) : (16 + kh + (e - 8)));
    int seg = k / 416, kr = k % 416;
    float v = 0.0f;
    if (n < NOUTC && seg < 3 && kr < 400) v = Wm[(size_t)(seg * 400 + kr) * NOUTC + n];
    d[e] = f2bf(v);
  }
}

__global__ void hw_zero(unsigned* __restrict__ p, size_t n){
  size_t i = (size_t)blockIdx.x * blockDim.x + threadIdx.x;
  size_t stride = (size_t)gridDim.x * blockDim.x;
  for (; i < n; i += stride) p[i] = 0u;
}

// Two-gate accumulation pass: acc{P,Q} (2 M-tiles each) over K, A from LDS, B from
// pre-swizzled global.  Kept small so only 32 acc VGPRs are live per pass.
static __device__ __forceinline__ void gate_pair_pass(
    const unsigned short* __restrict__ Ash, int apitch, int KT, int lane,
    const unsigned short* __restrict__ BP, const unsigned short* __restrict__ BQ,
    v8f accP[2], v8f accQ[2]){
  #pragma unroll 1
  for (int kt = 0; kt < KT; ++kt){
    v16bf a0 = load_afrag(Ash, apitch, 0,  kt*32, lane);
    v16bf a1 = load_afrag(Ash, apitch, 16, kt*32, lane);
    v16bf bp = load_bfrag(BP, kt, lane);
    accP[0] = wmma_bf16(a0, bp, accP[0]);
    accP[1] = wmma_bf16(a1, bp, accP[1]);
    v16bf bq = load_bfrag(BQ, kt, lane);
    accQ[0] = wmma_bf16(a0, bq, accQ[0]);
    accQ[1] = wmma_bf16(a1, bq, accQ[1]);
  }
}

// ---------------- Phase 1: attention LSTM (persistent, 1 workgroup, 25 waves) ----------------
// Wave w owns h-columns [16w,16w+16); gate tiles {j, j+400, j+800, j+1200} -> lane-local mix.
// Gate pairs (i,g) then (f,o) accumulated in two K passes to halve register pressure.
__global__ void __launch_bounds__(800)
hw_phase1(const float* __restrict__ strokes, const float* __restrict__ sentence,
          const unsigned short* __restrict__ Wc1, const float* __restrict__ b1,
          const float* __restrict__ Ww, const float* __restrict__ bw,
          unsigned short* __restrict__ in2b, unsigned short* __restrict__ in3b,
          unsigned short* __restrict__ xcat){
  extern __shared__ char smem[];
  unsigned short* Ash  = (unsigned short*)smem;                 // 32*480 bf16   (30720 B)
  float* Hsh   = (float*)(smem + 32*KP1*2);                     // 32*400 f32    (51200 B)
  float* psh   = (float*)((char*)Hsh   + 32*HH*4);              // 32*30         (3840 B)
  float* kap   = (float*)((char*)psh   + 32*30*4);              // 32*10         (1280 B)
  float* phish = (float*)((char*)kap   + 32*KK*4);              // 32*64         (8192 B)
  float* WwT   = (float*)((char*)phish + 32*UU*4);              // 30x400 (transposed, 48000 B)

  const int tid = threadIdx.x, lane = tid & 31, wave = tid >> 5;
  const int hi  = lane >> 4;
  const int col = wave * 16 + (lane & 15);   // owned h column, 0..399
  const int KT  = 15;
  // per-gate swizzled B base pointers (ntile = gate*25 + wave)
  const unsigned short* Bi = Wc1 + (size_t)(0*25 + wave) * KT * 512;
  const unsigned short* Bf = Wc1 + (size_t)(1*25 + wave) * KT * 512;
  const unsigned short* Bg = Wc1 + (size_t)(2*25 + wave) * KT * 512;
  const unsigned short* Bo = Wc1 + (size_t)(3*25 + wave) * KT * 512;

  for (int j = tid; j < 32*KP1; j += 800) Ash[j] = 0;           // x=w=h=0, pads=0
  for (int j = tid; j < 32*KK;  j += 800) kap[j] = 0.0f;
  for (int j = tid; j < HH*30;  j += 800){                       // WwT[q][k] = Ww[k][q]
    int q = j / HH, k = j % HH;
    WwT[j] = Ww[(size_t)k*30 + q];
  }
  __syncthreads();
  if (tid < 96){ int b = tid/3, i = tid%3;
    Ash[b*KP1 + i] = f2bf(strokes[(b*TT + 0)*3 + i]); }
  float cst[2][8];
  #pragma unroll
  for (int mt = 0; mt < 2; ++mt){
    #pragma unroll
    for (int r = 0; r < 8; ++r) cst[mt][r] = 0.0f;
  }
  __syncthreads();

  for (int t = 0; t < TT; ++t){
    // ---- pass A: gates i (cols +0) and g (cols +800) ----
    v8f accP[2], accQ[2];
    {
      float bi = b1[col], bg = b1[800 + col];
      #pragma unroll
      for (int r = 0; r < 8; ++r){ accP[0][r]=bi; accP[1][r]=bi; accQ[0][r]=bg; accQ[1][r]=bg; }
    }
    gate_pair_pass(Ash, KP1, KT, lane, Bi, Bg, accP, accQ);
    float t1[2][8];
    #pragma unroll
    for (int mt = 0; mt < 2; ++mt){
      #pragma unroll
      for (int r = 0; r < 8; ++r) t1[mt][r] = sigf(accP[mt][r]) * ftanh(accQ[mt][r]);
    }
    // ---- pass B: gates f (cols +400) and o (cols +1200) ----
    {
      float bfv = b1[400 + col], bo = b1[1200 + col];
      #pragma unroll
      for (int r = 0; r < 8; ++r){ accP[0][r]=bfv; accP[1][r]=bfv; accQ[0][r]=bo; accQ[1][r]=bo; }
    }
    gate_pair_pass(Ash, KP1, KT, lane, Bf, Bo, accP, accQ);
    __syncthreads();                         // all waves done reading A(t)
    #pragma unroll
    for (int mt = 0; mt < 2; ++mt){
      #pragma unroll
      for (int r = 0; r < 8; ++r){
        float c = sigf(accP[mt][r]) * cst[mt][r] + t1[mt][r];
        cst[mt][r] = c;
        float h = sigf(accQ[mt][r]) * ftanh(c);
        int row = mt*16 + hi*8 + r;
        unsigned short hb = f2bf(h);
        Hsh[row*HH + col]       = h;
        Ash[row*KP1 + 64 + col] = hb;                       // h for step t+1
        size_t gr = (size_t)t*BB + row;
        xcat[gr*KPM + col]       = hb;                      // lstm1 -> output GEMM
        in2b[gr*KP1 + 64 + col]  = hb;                      // lstm1 -> in2
      }
    }
    __syncthreads();                         // h(t) visible
    // p = h@Ww + bw, store exp(p); both operands stride-1 -> ds_load_b128 pairs
    for (int j = tid; j < 32*30; j += 800){
      int b = j/30, q = j%30;
      const float4* h4 = (const float4*)(Hsh + b*HH);
      const float4* w4 = (const float4*)(WwT + q*HH);
      float s = bw[q];
      #pragma unroll 4
      for (int k4 = 0; k4 < HH/4; ++k4){
        float4 ha = h4[k4], wa = w4[k4];
        s += ha.x*wa.x + ha.y*wa.y + ha.z*wa.z + ha.w*wa.w;
      }
      psh[j] = fexpn(s);
    }
    __syncthreads();
    if (tid < 32*KK){ int b = tid/KK, k = tid%KK; kap[tid] += psh[b*30 + 20 + k]; }
    __syncthreads();
    // phi[b,u] = sum_k alpha * exp(-beta*(kappa-u)^2)
    for (int j = tid; j < 32*UU; j += 800){
      int b = j >> 6, u = j & 63;
      float uf = (float)u, s = 0.0f;
      #pragma unroll
      for (int k = 0; k < KK; ++k){
        float al = psh[b*30 + k], be = psh[b*30 + 10 + k];
        float d = kap[b*KK + k] - uf;
        s += al * fexpn(-be*d*d);
      }
      phish[j] = s;
    }
    __syncthreads();
    // w[b,v] = sum_u phi * sentence
    for (int j = tid; j < 32*VV; j += 800){
      int b = j/VV, v = j%VV;
      float s = 0.0f;
      for (int u = 0; u < UU; ++u) s += phish[b*UU + u] * sentence[((size_t)b*UU + u)*VV + v];
      unsigned short wb = f2bf(s);
      Ash[b*KP1 + 3 + v] = wb;                              // xin(t+1)
      size_t gr = (size_t)t*BB + b;
      in2b[gr*KP1 + 3 + v] = wb;
      in3b[gr*KP1 + 3 + v] = wb;
    }
    if (tid < 96){
      int b = tid/3, i = tid%3;
      size_t gr = (size_t)t*BB + b;
      unsigned short xb = f2bf(strokes[(b*TT + t)*3 + i]);
      in2b[gr*KP1 + i] = xb;
      in3b[gr*KP1 + i] = xb;
      if (t + 1 < TT) Ash[b*KP1 + i] = f2bf(strokes[(b*TT + t + 1)*3 + i]);
    }
    __syncthreads();
  }
}

// ---------------- Phase 2/3: plain recurrent LSTM, input proj precomputed in Zbuf ----------------
__global__ void __launch_bounds__(800)
hw_phaseR(const unsigned short* __restrict__ Zbuf, const unsigned short* __restrict__ Usw,
          unsigned short* __restrict__ xcat, int xcoff, unsigned short* __restrict__ in3b){
  extern __shared__ char smem[];
  unsigned short* Ash = (unsigned short*)smem;               // 32*416 bf16
  const int tid = threadIdx.x, lane = tid & 31, wave = tid >> 5;
  const int hi  = lane >> 4;
  const int col = wave * 16 + (lane & 15);
  const int KT  = 13;
  const unsigned short* Bi = Usw + (size_t)(0*25 + wave) * KT * 512;
  const unsigned short* Bf = Usw + (size_t)(1*25 + wave) * KT * 512;
  const unsigned short* Bg = Usw + (size_t)(2*25 + wave) * KT * 512;
  const unsigned short* Bo = Usw + (size_t)(3*25 + wave) * KT * 512;

  for (int j = tid; j < 32*KPR; j += 800) Ash[j] = 0;
  float cst[2][8];
  #pragma unroll
  for (int mt = 0; mt < 2; ++mt){
    #pragma unroll
    for (int r = 0; r < 8; ++r) cst[mt][r] = 0.0f;
  }
  __syncthreads();

  for (int t = 0; t < TT; ++t){
    const unsigned short* Zt = Zbuf + (size_t)t*BB*NZ;
    v8f accP[2], accQ[2];
    // ---- pass A: i (+0) and g (+800), init from precomputed Z ----
    #pragma unroll
    for (int mt = 0; mt < 2; ++mt){
      #pragma unroll
      for (int r = 0; r < 8; ++r){
        int row = mt*16 + hi*8 + r;
        accP[mt][r] = bf2f(Zt[(size_t)row*NZ + col]);
        accQ[mt][r] = bf2f(Zt[(size_t)row*NZ + 800 + col]);
      }
    }
    gate_pair_pass(Ash, KPR, KT, lane, Bi, Bg, accP, accQ);
    float t1[2][8];
    #pragma unroll
    for (int mt = 0; mt < 2; ++mt){
      #pragma unroll
      for (int r = 0; r < 8; ++r) t1[mt][r] = sigf(accP[mt][r]) * ftanh(accQ[mt][r]);
    }
    // ---- pass B: f (+400) and o (+1200) ----
    #pragma unroll
    for (int mt = 0; mt < 2; ++mt){
      #pragma unroll
      for (int r = 0; r < 8; ++r){
        int row = mt*16 + hi*8 + r;
        accP[mt][r] = bf2f(Zt[(size_t)row*NZ + 400 + col]);
        accQ[mt][r] = bf2f(Zt[(size_t)row*NZ + 1200 + col]);
      }
    }
    gate_pair_pass(Ash, KPR, KT, lane, Bf, Bo, accP, accQ);
    __syncthreads();
    #pragma unroll
    for (int mt = 0; mt < 2; ++mt){
      #pragma unroll
      for (int r = 0; r < 8; ++r){
        float c = sigf(accP[mt][r]) * cst[mt][r] + t1[mt][r];
        cst[mt][r] = c;
        float h = sigf(accQ[mt][r]) * ftanh(c);
        int row = mt*16 + hi*8 + r;
        unsigned short hb = f2bf(h);
        Ash[row*KPR + col] = hb;
        size_t gr = (size_t)t*BB + row;
        xcat[gr*KPM + xcoff + col] = hb;
        if (in3b) in3b[gr*KP1 + 64 + col] = hb;
      }
    }
    __syncthreads();
  }
}

// ---------------- batched input-projection GEMM: Z = A(19200x480) @ Bsw(480x1600) + bias ----------------
__global__ void __launch_bounds__(256)
hw_gemm_proj(const unsigned short* __restrict__ Abuf, const unsigned short* __restrict__ Bsw,
             const float* __restrict__ bias, unsigned short* __restrict__ Zout){
  const int tid = threadIdx.x, lane = tid & 31, wave = tid >> 5;
  const int hi  = lane >> 4;
  const int mtile = blockIdx.y * 8 + wave;
  const int nt0   = blockIdx.x * 4;
  const int KT = 15;
  const unsigned short* Ab = Abuf + (size_t)mtile * 16 * KP1;
  v8f acc[4];
  #pragma unroll
  for (int q = 0; q < 4; ++q){
    float bv = bias[(nt0 + q)*16 + (lane & 15)];
    #pragma unroll
    for (int r = 0; r < 8; ++r) acc[q][r] = bv;
  }
  #pragma unroll 1
  for (int kt = 0; kt < KT; ++kt){
    v16bf a = load_afrag(Ab, KP1, 0, kt*32, lane);
    #pragma unroll
    for (int q = 0; q < 4; ++q){
      v16bf bf = load_bfrag(Bsw, (size_t)(nt0 + q)*KT + kt, lane);
      acc[q] = wmma_bf16(a, bf, acc[q]);
    }
  }
  #pragma unroll
  for (int q = 0; q < 4; ++q){
    int colq = (nt0 + q)*16 + (lane & 15);
    #pragma unroll
    for (int r = 0; r < 8; ++r){
      int row = mtile*16 + hi*8 + r;
      Zout[(size_t)row*NZ + colq] = f2bf(acc[q][r]);
    }
  }
}

// ---------------- output GEMM: y = Xcat(19200x1248) @ Wmsw(1248x128) ----------------
__global__ void __launch_bounds__(256)
hw_gemm_out(const unsigned short* __restrict__ Xcat, const unsigned short* __restrict__ Wmsw,
            float* __restrict__ ybuf){
  const int tid = threadIdx.x, lane = tid & 31, wave = tid >> 5;
  const int hi  = lane >> 4;
  const int mtile = blockIdx.x * 8 + wave;
  const int KT = 39;
  const unsigned short* Ab = Xcat + (size_t)mtile * 16 * KPM;
  v8f acc[8];
  #pragma unroll
  for (int q = 0; q < 8; ++q){
    #pragma unroll
    for (int r = 0; r < 8; ++r) acc[q][r] = 0.0f;
  }
  #pragma unroll 1
  for (int kt = 0; kt < KT; ++kt){
    v16bf a = load_afrag(Ab, KPM, 0, kt*32, lane);
    #pragma unroll
    for (int q = 0; q < 8; ++q){
      v16bf bf = load_bfrag(Wmsw, (size_t)q*KT + kt, lane);
      acc[q] = wmma_bf16(a, bf, acc[q]);
    }
  }
  #pragma unroll
  for (int q = 0; q < 8; ++q){
    int colq = q*16 + (lane & 15);
    #pragma unroll
    for (int r = 0; r < 8; ++r){
      int row = mtile*16 + hi*8 + r;
      ybuf[(size_t)row*NYM + colq] = acc[q][r];
    }
  }
}

// ---------------- nonlinearities + (T,B)->(B,T) transpose (precise libm, one-shot) ----------------
__global__ void hw_nonlin(const float* __restrict__ ybuf, const float* __restrict__ bm,
                          float* __restrict__ out){
  int r = blockIdx.x * 256 + threadIdx.x;
  if (r >= MROWS) return;
  const float* y = ybuf + (size_t)r * NYM;
  int t = r / BB, b = r % BB;
  float* o = out + ((size_t)b*TT + t) * NOUTC;
  o[0] = 1.0f/(1.0f + expf(-(y[0] + bm[0])));
  float v[20]; float mx = -3.4e38f;
  #pragma unroll
  for (int i = 0; i < 20; ++i){ v[i] = y[1+i] + bm[1+i]; mx = fmaxf(mx, v[i]); }
  float s = 0.0f;
  #pragma unroll
  for (int i = 0; i < 20; ++i){ v[i] = expf(v[i] - mx); s += v[i]; }
  float inv = 1.0f/s;
  #pragma unroll
  for (int i = 0; i < 20; ++i) o[1+i] = v[i]*inv;
  #pragma unroll
  for (int i = 0; i < 40; ++i) o[21+i] = y[21+i] + bm[21+i];      // mu1, mu2
  #pragma unroll
  for (int i = 0; i < 40; ++i) o[61+i] = expf(y[61+i] + bm[61+i]); // s1, s2
  #pragma unroll
  for (int i = 0; i < 20; ++i) o[101+i] = tanhf(y[101+i] + bm[101+i]); // rho
}

extern "C" void kernel_launch(void* const* d_in, const int* in_sizes, int n_in,
                              void* d_out, int out_size, void* d_ws, size_t ws_size,
                              hipStream_t stream){
  const float* strokes  = (const float*)d_in[0];
  const float* sentence = (const float*)d_in[1];
  const float* W1  = (const float*)d_in[2];
  const float* U1r = (const float*)d_in[3];
  const float* b1  = (const float*)d_in[4];
  const float* Ww  = (const float*)d_in[5];
  const float* bw  = (const float*)d_in[6];
  const float* W2  = (const float*)d_in[7];
  const float* U2r = (const float*)d_in[8];
  const float* b2  = (const float*)d_in[9];
  const float* W3  = (const float*)d_in[10];
  const float* U3r = (const float*)d_in[11];
  const float* b3  = (const float*)d_in[12];
  const float* Wm  = (const float*)d_in[13];
  const float* bm  = (const float*)d_in[14];
  float* out = (float*)d_out;

  char* ws = (char*)d_ws;
  size_t off = 0;
  auto carve = [&](size_t bytes)->char*{
    char* p = ws + off;
    off += (bytes + 255) & ~(size_t)255;
    return p;
  };
  unsigned short* W1sw = (unsigned short*)carve((size_t)15*100*1024);   // 480x1600 combined [W1;U1r;0]
  unsigned short* U2sw = (unsigned short*)carve((size_t)13*100*1024);   // 416x1600
  unsigned short* U3sw = (unsigned short*)carve((size_t)13*100*1024);
  unsigned short* W2sw = (unsigned short*)carve((size_t)15*100*1024);   // 480x1600 [W2;0]
  unsigned short* W3sw = (unsigned short*)carve((size_t)15*100*1024);
  unsigned short* Wmsw = (unsigned short*)carve((size_t)39*8*1024);     // 1248x128
  unsigned short* in2b = (unsigned short*)carve((size_t)MROWS*KP1*2);
  unsigned short* in3b = (unsigned short*)carve((size_t)MROWS*KP1*2);
  unsigned short* xcat = (unsigned short*)carve((size_t)MROWS*KPM*2);
  unsigned short* zbuf = (unsigned short*)carve((size_t)MROWS*NZ*2);
  float*          ybuf = (float*)carve((size_t)MROWS*NYM*4);

  // 1) Weight conversion / swizzle (bf16, WMMA B-fragment order)
  int th15 = 15*100*32, th13 = 13*100*32, thM = 39*8*32;
  hw_swizzleB<<<(th15+255)/256, 256, 0, stream>>>(W1sw, W1, 64, U1r, 400, NZ, 15, 100);
  hw_swizzleB<<<(th13+255)/256, 256, 0, stream>>>(U2sw, U2r, 400, nullptr, 0, NZ, 13, 100);
  hw_swizzleB<<<(th13+255)/256, 256, 0, stream>>>(U3sw, U3r, 400, nullptr, 0, NZ, 13, 100);
  hw_swizzleB<<<(th15+255)/256, 256, 0, stream>>>(W2sw, W2, 464, nullptr, 0, NZ, 15, 100);
  hw_swizzleB<<<(th15+255)/256, 256, 0, stream>>>(W3sw, W3, 464, nullptr, 0, NZ, 15, 100);
  hw_swizzleWm<<<(thM+255)/256, 256, 0, stream>>>(Wmsw, Wm);

  // 2) Zero activation buffers (pads must be zero; harness does not re-poison between replays)
  hw_zero<<<2048, 256, 0, stream>>>((unsigned*)in2b, (size_t)MROWS*KP1/2);
  hw_zero<<<2048, 256, 0, stream>>>((unsigned*)in3b, (size_t)MROWS*KP1/2);
  hw_zero<<<2048, 256, 0, stream>>>((unsigned*)xcat, (size_t)MROWS*KPM/2);

  // 3) Phase 1: attention LSTM scan (persistent single workgroup, 25 waves)
  size_t shm1 = 32*KP1*2 + 32*HH*4 + 32*30*4 + 32*KK*4 + 32*UU*4 + HH*30*4; // 143232 B
  hw_phase1<<<1, 800, shm1, stream>>>(strokes, sentence, W1sw, b1, Ww, bw, in2b, in3b, xcat);

  // 4) Layer-2 input projection (full-GPU GEMM), then recurrent scan
  hw_gemm_proj<<<dim3(25,150), 256, 0, stream>>>(in2b, W2sw, b2, zbuf);
  hw_phaseR<<<1, 800, (size_t)(32*KPR*2), stream>>>(zbuf, U2sw, xcat, 416, in3b);

  // 5) Layer-3 input projection, then recurrent scan
  hw_gemm_proj<<<dim3(25,150), 256, 0, stream>>>(in3b, W3sw, b3, zbuf);
  hw_phaseR<<<1, 800, (size_t)(32*KPR*2), stream>>>(zbuf, U3sw, xcat, 832, nullptr);

  // 6) Output projection + mixture-density nonlinearities
  hw_gemm_out<<<150, 256, 0, stream>>>(xcat, Wmsw, ybuf);
  hw_nonlin<<<(MROWS+255)/256, 256, 0, stream>>>(ybuf, bm, out);
}
